// transformer_42502996361353
// MI455X (gfx1250) — matrix operand order repaired and checked by reading, not compiled, not run on hardware
//
#include <hip/hip_runtime.h>
#include <hip/hip_bf16.h>
#include <math.h>

// Problem dims (compile-time constants from the reference)
#define Lg  6
#define Hg  8
#define Dg  512
#define Fg  2048
#define Sg  512
#define Vg  50257
#define Bg  8
#define HDg 64
#define NTg (Bg*Sg)   // 4096 tokens

typedef _Float16 v16h __attribute__((ext_vector_type(16)));
typedef _Float16 v8h  __attribute__((ext_vector_type(8)));
typedef _Float16 v4h  __attribute__((ext_vector_type(4)));
typedef float    v8f  __attribute__((ext_vector_type(8)));
typedef int      v4i  __attribute__((ext_vector_type(4)));

// ---------------------------------------------------------------------------
// Async global->LDS staging (CDNA5). Guarded so compile never breaks.
// Builtin signature (probe-discovered): (v4i AS1*, v4i AS3*, imm offset, imm cpol)
// ---------------------------------------------------------------------------
#if defined(__AMDGCN__) && \
    __has_builtin(__builtin_amdgcn_global_load_async_to_lds_b128) && \
    __has_builtin(__builtin_amdgcn_s_wait_asynccnt)
#define ASYNC_LDS 1
#else
#define ASYNC_LDS 0
#endif

__device__ __forceinline__ void async_copy16(const _Float16* g, _Float16* l)
{
#if ASYNC_LDS
    __builtin_amdgcn_global_load_async_to_lds_b128(
        (__attribute__((address_space(1))) v4i*)const_cast<_Float16*>(g),
        (__attribute__((address_space(3))) v4i*)l, 0, 0);
#else
    *(v8h*)l = *(const v8h*)g;
#endif
}

__device__ __forceinline__ void wait_async()
{
#if ASYNC_LDS
    __builtin_amdgcn_s_wait_asynccnt(0);
#endif
}

// ---------------------------------------------------------------------------
// fp32 -> f16 flat convert (vectorized x4)
// ---------------------------------------------------------------------------
__global__ __launch_bounds__(256) void cvt_kernel(
    const float* __restrict__ in, _Float16* __restrict__ out, long long n)
{
    long long i = ((long long)blockIdx.x * 256 + threadIdx.x) * 4;
    if (i < n) {
        float4 f = *(const float4*)(in + i);
        v4h o = { (_Float16)f.x, (_Float16)f.y, (_Float16)f.z, (_Float16)f.w };
        *(v4h*)(out + i) = o;
    }
}

// ---------------------------------------------------------------------------
// Tiled transpose + convert-to-f16: in [R,C] (type T) -> out [C,R] f16
// grid (C/64, R/64, Z), z-stride R*C on both sides. 256 threads.
// ---------------------------------------------------------------------------
template<typename T>
__global__ __launch_bounds__(256) void transpose_cvt_kernel(
    const T* __restrict__ in, _Float16* __restrict__ out, int R, int C)
{
    __shared__ _Float16 tile[64][65];
    long long zofs = (long long)blockIdx.z * R * C;
    in  += zofs;
    out += zofs;
    int c0 = blockIdx.x * 64, r0 = blockIdx.y * 64;
    int tx = threadIdx.x & 63, ty = threadIdx.x >> 6;   // 64 x 4
    #pragma unroll
    for (int i = 0; i < 64; i += 4)
        tile[ty + i][tx] = (_Float16)in[(long long)(r0 + ty + i) * C + (c0 + tx)];
    __syncthreads();
    #pragma unroll
    for (int i = 0; i < 64; i += 4)
        out[(long long)(c0 + ty + i) * R + (r0 + tx)] = tile[tx][ty + i];
}

// ---------------------------------------------------------------------------
// Embedding: h[b,s,:] = token_emb[x[b,s]] + pos_emb[s]   (fp32 residual stream)
// ---------------------------------------------------------------------------
__global__ __launch_bounds__(256) void embed_kernel(
    const int* __restrict__ x, const float* __restrict__ emb,
    const float* __restrict__ pos, float* __restrict__ h)
{
    int idx = blockIdx.x * 256 + threadIdx.x;   // over NTg*Dg
    int t = idx / Dg, d = idx % Dg;
    int s = t % Sg;
    int tok = x[t];
    h[idx] = emb[tok * Dg + d] + pos[s * Dg + d];
}

// ---------------------------------------------------------------------------
// LayerNorm (fp32 in -> f16 out): one block (256 threads) per token, D=512
// ---------------------------------------------------------------------------
__global__ __launch_bounds__(256) void ln_kernel(
    const float* __restrict__ in, const float* __restrict__ g,
    const float* __restrict__ b, _Float16* __restrict__ out)
{
    __shared__ float red[256];
    int t = blockIdx.x;
    int tid = threadIdx.x;
    const float* row = in + (long long)t * Dg;
    float v0 = row[tid], v1 = row[tid + 256];

    red[tid] = v0 + v1; __syncthreads();
    for (int s = 128; s > 0; s >>= 1) {
        if (tid < s) red[tid] += red[tid + s];
        __syncthreads();
    }
    float mean = red[0] * (1.0f / Dg);
    __syncthreads();

    float d0 = v0 - mean, d1 = v1 - mean;
    red[tid] = d0 * d0 + d1 * d1; __syncthreads();
    for (int s = 128; s > 0; s >>= 1) {
        if (tid < s) red[tid] += red[tid + s];
        __syncthreads();
    }
    float rstd = rsqrtf(red[0] * (1.0f / Dg) + 1e-5f);

    _Float16* orow = out + (long long)t * Dg;
    orow[tid]       = (_Float16)(d0 * rstd * g[tid]       + b[tid]);
    orow[tid + 256] = (_Float16)(d1 * rstd * g[tid + 256] + b[tid + 256]);
}

// ---------------------------------------------------------------------------
// WMMA GEMM: C[z] = epilogue( A[z] @ B[z]^T-layout )
//   A: f16 [M,K] row-major (lda), B: f16 [N,K] row-major (ldb) -> K contiguous
//   256 threads = 8 wave32s; macro tile 128(M) x 128(N); K step 32
//   wave tile 32x64 = 2x4 v_wmma_f32_16x16x32_f16; fp32 accumulate
//   double-buffered LDS, async global->LDS staging
//   NBOUND: enable N-edge handling (only ctx N=64 and logits N=50257 need it)
//   epilogue: *scale, +bias[col], gelu, causal mask, +residual(fp32), cast OutT
// ---------------------------------------------------------------------------
template<bool BIAS, bool GELU, bool RES, bool CAUSAL, bool NBOUND, typename OutT>
__global__ __launch_bounds__(256) void gemm_wmma_kernel(
    const _Float16* __restrict__ A, int lda, long long sA,
    const _Float16* __restrict__ B, int ldb, long long sB,
    const float* __restrict__ bias,
    const float* __restrict__ res,
    OutT* __restrict__ C, int ldc, long long sC,
    int N, int K, float scale)
{
    __shared__ __align__(16) _Float16 As[2][128][40];  // 32 K + pad (80B rows, 16B aligned)
    __shared__ __align__(16) _Float16 Bs[2][128][40];  // N-major, K contiguous

    A += (long long)blockIdx.z * sA;
    B += (long long)blockIdx.z * sB;
    C += (long long)blockIdx.z * sC;

    const int tid   = threadIdx.x;
    const int lane  = tid & 31;
    const int wave  = tid >> 5;
    const int wm    = (wave >> 1) * 32;   // 4 wave-rows
    const int wn    = (wave & 1) * 64;    // 2 wave-cols
    const int lrow  = lane & 15;
    const int lhalf = lane >> 4;

    const int blockM = blockIdx.y * 128;
    const int blockN = blockIdx.x * 128;

    // staging assignment: thread -> (row, 16-half segment); 2x b128 per matrix
    const int srow = tid >> 1;
    const int sseg = (tid & 1) * 16;
    const bool bvalid = !NBOUND || ((blockN + srow) < N);

    v8f acc[2][4] = {};

    const int nk = K >> 5;

    // ---- prologue: stage k-tile 0 into buffer 0
    {
        const _Float16* ga = A + (long long)(blockM + srow) * lda + sseg;
        async_copy16(ga,     &As[0][srow][sseg]);
        async_copy16(ga + 8, &As[0][srow][sseg + 8]);
        if (bvalid) {
            const _Float16* gb = B + (long long)(blockN + srow) * ldb + sseg;
            async_copy16(gb,     &Bs[0][srow][sseg]);
            async_copy16(gb + 8, &Bs[0][srow][sseg + 8]);
        } else {
            v8h z = {};
            *(v8h*)&Bs[0][srow][sseg]     = z;
            *(v8h*)&Bs[0][srow][sseg + 8] = z;
        }
    }
    wait_async();
    __syncthreads();

    for (int kt = 0; kt < nk; ++kt) {
        const int cur = kt & 1;

        // ---- stage next k-tile into the other buffer (overlaps WMMA below)
        if (kt + 1 < nk) {
            const int nxt = cur ^ 1;
            const int k0 = (kt + 1) << 5;
            const _Float16* ga = A + (long long)(blockM + srow) * lda + k0 + sseg;
            async_copy16(ga,     &As[nxt][srow][sseg]);
            async_copy16(ga + 8, &As[nxt][srow][sseg + 8]);
            if (bvalid) {
                const _Float16* gb = B + (long long)(blockN + srow) * ldb + k0 + sseg;
                async_copy16(gb,     &Bs[nxt][srow][sseg]);
                async_copy16(gb + 8, &Bs[nxt][srow][sseg + 8]);
            } else {
                v8h z = {};
                *(v8h*)&Bs[nxt][srow][sseg]     = z;
                *(v8h*)&Bs[nxt][srow][sseg + 8] = z;
            }
        }

        // ---- fragments per ISA 7.12.2 (wave32):
        // A: lane = row M=lane%16; K = {kb..kb+7, kb+16..kb+23}, kb = 8*(lane/16)
        // B: lane = col N=lane%16; K = {kb..kb+7, kb+16..kb+23}, kb = 8*(lane/16)
        const int kb  = lhalf * 8;
        const int klo = lhalf * 16;
        v16h afrag[2], bfrag[4];
        #pragma unroll
        for (int i = 0; i < 2; ++i) {
            const _Float16* pa = &As[cur][wm + i * 16 + lrow][kb];
            v8h lo = *(const v8h*)pa;
            v8h hi = *(const v8h*)(pa + 16);
            afrag[i] = __builtin_shufflevector(lo, hi,
                         0,1,2,3,4,5,6,7,8,9,10,11,12,13,14,15);
        }
        #pragma unroll
        for (int j = 0; j < 4; ++j) {
            const _Float16* pb = &Bs[cur][wn + j * 16 + lrow][kb];
            v8h lo = *(const v8h*)pb;
            v8h hi = *(const v8h*)(pb + 16);
            bfrag[j] = __builtin_shufflevector(lo, hi,
                         0,1,2,3,4,5,6,7,8,9,10,11,12,13,14,15);
        }

        #pragma unroll
        for (int i = 0; i < 2; ++i)
            #pragma unroll
            for (int j = 0; j < 4; ++j)
                acc[i][j] = __builtin_amdgcn_wmma_f32_16x16x32_f16(
                    false, afrag[i], false, bfrag[j],
                    (short)0, acc[i][j], false, false);

        // next-tile data ready + everyone done reading cur before it is reused
        wait_async();
        __syncthreads();
    }

    // ---- epilogue; C/D layout: VGPR r -> row r + 8*(lane/16), col = lane%16
    #pragma unroll
    for (int i = 0; i < 2; ++i) {
        #pragma unroll
        for (int j = 0; j < 4; ++j) {
            int col = blockN + wn + j * 16 + lrow;
            if (!NBOUND || col < N) {
                #pragma unroll
                for (int r = 0; r < 8; ++r) {
                    int rowi = blockM + wm + i * 16 + r + lhalf * 8;
                    float cv = acc[i][j][r] * scale;
                    if (BIAS)   cv += bias[col];
                    if (GELU)   cv = 0.5f * cv * (1.0f + erff(cv * 0.70710678f));
                    if (CAUSAL) { if (col > rowi) cv = -1e30f; }
                    if (RES)    cv += res[(long long)rowi * ldc + col];
                    C[(long long)rowi * ldc + col] = (OutT)cv;
                }
            }
        }
    }
}

// ---------------------------------------------------------------------------
// Row softmax over S=512: fp32 scores in -> f16 attention out
// grid = (S, H); layout [H][S][S]
// ---------------------------------------------------------------------------
__global__ __launch_bounds__(256) void softmax_kernel(
    const float* __restrict__ sc, _Float16* __restrict__ att)
{
    __shared__ float red[256];
    int m = blockIdx.x, h = blockIdx.y, tid = threadIdx.x;
    long long base = ((long long)h * Sg + m) * Sg;
    const float* row = sc + base;
    float a = row[tid], b = row[tid + 256];

    red[tid] = fmaxf(a, b); __syncthreads();
    for (int s = 128; s > 0; s >>= 1) {
        if (tid < s) red[tid] = fmaxf(red[tid], red[tid + s]);
        __syncthreads();
    }
    float mx = red[0];
    __syncthreads();

    float e0 = expf(a - mx), e1 = expf(b - mx);
    red[tid] = e0 + e1; __syncthreads();
    for (int s = 128; s > 0; s >>= 1) {
        if (tid < s) red[tid] += red[tid + s];
        __syncthreads();
    }
    float inv = 1.0f / red[0];
    _Float16* orow = att + base;
    orow[tid]       = (_Float16)(e0 * inv);
    orow[tid + 256] = (_Float16)(e1 * inv);
}

// ---------------------------------------------------------------------------
// Cross-entropy: one block per token; loss += (lse - logit[target]) / NT
// ---------------------------------------------------------------------------
__global__ void loss_init_kernel(float* loss) { *loss = 0.0f; }

__global__ __launch_bounds__(256) void loss_kernel(
    const float* __restrict__ logits, const int* __restrict__ targets,
    float* __restrict__ loss)
{
    __shared__ float red[256];
    int m = blockIdx.x, tid = threadIdx.x;
    const float* row = logits + (long long)m * Vg;

    float mx = -3.0e38f;
    for (int i = tid; i < Vg; i += 256) mx = fmaxf(mx, row[i]);
    red[tid] = mx; __syncthreads();
    for (int s = 128; s > 0; s >>= 1) {
        if (tid < s) red[tid] = fmaxf(red[tid], red[tid + s]);
        __syncthreads();
    }
    mx = red[0];
    __syncthreads();

    float sum = 0.0f;
    for (int i = tid; i < Vg; i += 256) sum += expf(row[i] - mx);
    red[tid] = sum; __syncthreads();
    for (int s = 128; s > 0; s >>= 1) {
        if (tid < s) red[tid] += red[tid + s];
        __syncthreads();
    }
    if (tid == 0) {
        float lse = mx + logf(red[0]);
        int tgt = targets[m];
        atomicAdd(loss, (lse - row[tgt]) * (1.0f / (float)NTg));
    }
}

// ---------------------------------------------------------------------------
// Host orchestration
// ---------------------------------------------------------------------------
extern "C" void kernel_launch(void* const* d_in, const int* in_sizes, int n_in,
                              void* d_out, int out_size, void* d_ws, size_t ws_size,
                              hipStream_t stream)
{
    (void)in_sizes; (void)n_in; (void)out_size; (void)ws_size;

    const int*   x       = (const int*)d_in[0];
    const int*   targets = (const int*)d_in[1];
    const float* tokemb  = (const float*)d_in[2];
    const float* posemb  = (const float*)d_in[3];
    const float* ln1g    = (const float*)d_in[4];
    const float* ln1b    = (const float*)d_in[5];
    const float* Wq      = (const float*)d_in[6];
    const float* bq      = (const float*)d_in[7];
    const float* Wk      = (const float*)d_in[8];
    const float* bk      = (const float*)d_in[9];
    const float* Wv      = (const float*)d_in[10];
    const float* bv      = (const float*)d_in[11];
    const float* Wo      = (const float*)d_in[12];
    const float* bo      = (const float*)d_in[13];
    const float* ln2g    = (const float*)d_in[14];
    const float* ln2b    = (const float*)d_in[15];
    const float* W1      = (const float*)d_in[16];
    const float* b1      = (const float*)d_in[17];
    const float* W2      = (const float*)d_in[18];
    const float* b2      = (const float*)d_in[19];
    const float* lnfg    = (const float*)d_in[20];
    const float* lnfb    = (const float*)d_in[21];

    float* logits = (float*)d_out;
    float* loss   = logits + (long long)NTg * Vg;

    // ---- workspace carve-out
    char* p = (char*)d_ws;
    auto alloc = [&](long long bytes) {
        char* r = p;
        p += (bytes + 255) & ~255ll;
        return (void*)r;
    };
    // fp32
    float* h  = (float*)alloc((long long)NTg * Dg * 4);
    float* sc = (float*)alloc((long long)Hg * Sg * Sg * 4);
    // f16 activations
    _Float16* xnh  = (_Float16*)alloc((long long)NTg * Dg * 2);
    _Float16* qh   = (_Float16*)alloc((long long)NTg * Dg * 2);
    _Float16* khh  = (_Float16*)alloc((long long)NTg * Dg * 2);
    _Float16* vh   = (_Float16*)alloc((long long)NTg * Dg * 2);
    _Float16* vTh  = (_Float16*)alloc((long long)NTg * Dg * 2);
    _Float16* ctxh = (_Float16*)alloc((long long)NTg * Dg * 2);
    _Float16* ynh  = (_Float16*)alloc((long long)NTg * Dg * 2);
    _Float16* ffah = (_Float16*)alloc((long long)NTg * Fg * 2);
    _Float16* atth = (_Float16*)alloc((long long)Hg * Sg * Sg * 2);
    // f16 transposed weights [N,K] + f16 embedding [V,D]
    _Float16* wqT  = (_Float16*)alloc((long long)Lg * Dg * Dg * 2);
    _Float16* wkT  = (_Float16*)alloc((long long)Lg * Dg * Dg * 2);
    _Float16* wvT  = (_Float16*)alloc((long long)Lg * Dg * Dg * 2);
    _Float16* woT  = (_Float16*)alloc((long long)Lg * Dg * Dg * 2);
    _Float16* w1T  = (_Float16*)alloc((long long)Lg * Dg * Fg * 2);
    _Float16* w2T  = (_Float16*)alloc((long long)Lg * Fg * Dg * 2);
    _Float16* embh = (_Float16*)alloc((long long)Vg * Dg * 2);

    const dim3 blk(256);
    const float attn_scale = 1.0f / 8.0f;   // 1/sqrt(HD)

    // ---- one-time (per launch) weight conversion, batched over layers via z
    {
        dim3 gdd(Dg / 64, Dg / 64, Lg);
        transpose_cvt_kernel<float><<<gdd, blk, 0, stream>>>(Wq, wqT, Dg, Dg);
        transpose_cvt_kernel<float><<<gdd, blk, 0, stream>>>(Wk, wkT, Dg, Dg);
        transpose_cvt_kernel<float><<<gdd, blk, 0, stream>>>(Wv, wvT, Dg, Dg);
        transpose_cvt_kernel<float><<<gdd, blk, 0, stream>>>(Wo, woT, Dg, Dg);
        dim3 gdf(Fg / 64, Dg / 64, Lg);   // [D,F] -> [F,D]
        transpose_cvt_kernel<float><<<gdf, blk, 0, stream>>>(W1, w1T, Dg, Fg);
        dim3 gfd(Dg / 64, Fg / 64, Lg);   // [F,D] -> [D,F]
        transpose_cvt_kernel<float><<<gfd, blk, 0, stream>>>(W2, w2T, Fg, Dg);
        long long ne = (long long)Vg * Dg;
        cvt_kernel<<<(unsigned)((ne / 4 + 255) / 256), blk, 0, stream>>>(tokemb, embh, ne);
    }

    embed_kernel<<<(NTg * Dg) / 256, blk, 0, stream>>>(x, tokemb, posemb, h);

    for (int l = 0; l < Lg; ++l) {
        ln_kernel<<<NTg, blk, 0, stream>>>(h, ln1g + l * Dg, ln1b + l * Dg, xnh);

        // q/k/v = xn @ W + b   (f16 out, N=512 -> no edge handling)
        dim3 gqkv(Dg / 128, NTg / 128, 1);
        gemm_wmma_kernel<true, false, false, false, false, _Float16><<<gqkv, blk, 0, stream>>>(
            xnh, Dg, 0, wqT + (long long)l * Dg * Dg, Dg, 0, bq + l * Dg, nullptr,
            qh, Dg, 0, Dg, Dg, 1.0f);
        gemm_wmma_kernel<true, false, false, false, false, _Float16><<<gqkv, blk, 0, stream>>>(
            xnh, Dg, 0, wkT + (long long)l * Dg * Dg, Dg, 0, bk + l * Dg, nullptr,
            khh, Dg, 0, Dg, Dg, 1.0f);
        gemm_wmma_kernel<true, false, false, false, false, _Float16><<<gqkv, blk, 0, stream>>>(
            xnh, Dg, 0, wvT + (long long)l * Dg * Dg, Dg, 0, bv + l * Dg, nullptr,
            vh, Dg, 0, Dg, Dg, 1.0f);

        // vT[b][n=h*HD+d][s] = v[b][s][n]  (per-batch 512x512 transpose, z=B)
        dim3 gvt(Dg / 64, Sg / 64, Bg);
        transpose_cvt_kernel<_Float16><<<gvt, blk, 0, stream>>>(vh, vTh, Sg, Dg);

        for (int b = 0; b < Bg; ++b) {
            const _Float16* qb = qh  + (long long)b * Sg * Dg;
            const _Float16* kb = khh + (long long)b * Sg * Dg;
            const _Float16* vtb = vTh + (long long)b * Dg * Sg;
            _Float16*       cb = ctxh + (long long)b * Sg * Dg;

            // scores[h] = (Q_h @ K_h^T)/8, causal mask; per-head z-stride HD
            dim3 gs(Sg / 128, Sg / 128, Hg);
            gemm_wmma_kernel<false, false, false, true, false, float><<<gs, blk, 0, stream>>>(
                qb, Dg, HDg, kb, Dg, HDg, nullptr, nullptr,
                sc, Sg, (long long)Sg * Sg, Sg, HDg, attn_scale);

            dim3 gsm(Sg, Hg);
            softmax_kernel<<<gsm, blk, 0, stream>>>(sc, atth);

            // ctx_h = att_h @ V_h    (B operand = V^T rows [HD,S]; N=64 -> edge)
            dim3 gc(1, Sg / 128, Hg);
            gemm_wmma_kernel<false, false, false, false, true, _Float16><<<gc, blk, 0, stream>>>(
                atth, Sg, (long long)Sg * Sg, vtb, Sg, (long long)HDg * Sg, nullptr, nullptr,
                cb, Dg, HDg, HDg, Sg, 1.0f);
        }

        // h = h + ctx @ Wo + bo   (fp32 out, residual)
        dim3 gp(Dg / 128, NTg / 128, 1);
        gemm_wmma_kernel<true, false, true, false, false, float><<<gp, blk, 0, stream>>>(
            ctxh, Dg, 0, woT + (long long)l * Dg * Dg, Dg, 0, bo + l * Dg, h,
            h, Dg, 0, Dg, Dg, 1.0f);

        ln_kernel<<<NTg, blk, 0, stream>>>(h, ln2g + l * Dg, ln2b + l * Dg, ynh);

        // ffa = gelu(yn @ W1 + b1)   (f16 out)
        dim3 g1(Fg / 128, NTg / 128, 1);
        gemm_wmma_kernel<true, true, false, false, false, _Float16><<<g1, blk, 0, stream>>>(
            ynh, Dg, 0, w1T + (long long)l * Dg * Fg, Dg, 0, b1 + l * Fg, nullptr,
            ffah, Fg, 0, Fg, Dg, 1.0f);

        // h = h + ffa @ W2 + b2   (fp32 out, residual)
        dim3 g2(Dg / 128, NTg / 128, 1);
        gemm_wmma_kernel<true, false, true, false, false, float><<<g2, blk, 0, stream>>>(
            ffah, Fg, 0, w2T + (long long)l * Fg * Dg, Fg, 0, b2 + l * Dg, h,
            h, Dg, 0, Dg, Fg, 1.0f);
    }

    // final LN + tied-weight vocab projection: logits = ln(h) @ token_emb^T
    ln_kernel<<<NTg, blk, 0, stream>>>(h, lnfg, lnfb, xnh);
    dim3 gl((Vg + 127) / 128, NTg / 128, 1);
    gemm_wmma_kernel<false, false, false, false, true, float><<<gl, blk, 0, stream>>>(
        xnh, Dg, 0, embh, Dg, 0, nullptr, nullptr,
        logits, Vg, 0, Vg, Dg, 1.0f);

    loss_init_kernel<<<1, 1, 0, stream>>>(loss);
    loss_kernel<<<NTg, blk, 0, stream>>>(logits, targets, loss);
}
